// SRVC_19585050870075
// MI455X (gfx1250) — compile-verified, matrix operand order checked by reading, not run on hardware
//
#include <hip/hip_runtime.h>
#include <hip/hip_bf16.h>

typedef __attribute__((ext_vector_type(16))) _Float16 v16h;
typedef __attribute__((ext_vector_type(8)))  _Float16 v8h;
typedef __attribute__((ext_vector_type(8)))  float    v8f;
typedef __attribute__((ext_vector_type(4))) unsigned int u32x4;
typedef __attribute__((ext_vector_type(8))) int          i32x8;
typedef __attribute__((ext_vector_type(4))) int          i32x4;

// ---------------- problem geometry (compile-time) ----------------
#define NB   4          // batch
#define H0   360
#define W0   640
#define GH   52         // 364/7
#define GW   92         // 644/7
#define HP   364
#define WP   644
#define HWp  (HP*WP)    // 234416
#define BP   (NB*GH*GW) // 19136 patches
#define N1   (BP*25)    // conv1 output pixels
#define N2   (BP*9)     // conv2 output pixels
#define N3   (BP)       // kern/bias GEMM
#define N56  (NB*HWp)   // conv3/conv4 output pixels  937664

__device__ __forceinline__ float rgb_mean(int c) {
    return c == 0 ? 114.444f : (c == 1 ? 111.4605f : 103.02f);
}

// ---------------- GEMM parameter block ----------------
struct GP {
    const _Float16* A;     // f16 weights [Mp][Kp]
    const _Float16* Bsrc;  // f16 activation source for im2col gather
    _Float16*       outH;  // f16 output (stages 1,2,3,5)
    float*          outF;  // f32 final output (stage 6)
    const float*    bias0;
    const float*    bias1; // stage 3 second bias (bb)
    int Mreal, Mp, Kp, N;
};

// ---------------- per-n decode (hoisted out of K loop) ----------------
template<int STAGE>
__device__ __forceinline__ void decodeN(const GP& g, int n,
                                        int& i0, int& i1, int& i2, bool& valid) {
    valid = n < g.N;
    int nn = valid ? n : 0;
    i1 = 0; i2 = 0;
    if constexpr (STAGE == 1) {
        int p = nn / 25, pix = nn % 25;
        i0 = p * 243 + (pix / 5 + 1) * 9 + (pix % 5 + 1);
    } else if constexpr (STAGE == 2) {
        int p = nn / 9, pix = nn % 9;
        i0 = p * 1600 + (pix / 3) * 5 + (pix % 3);     // p*64*25 + oy*5 + ox
    } else if constexpr (STAGE == 3) {
        i0 = nn * 576;
    } else {                                           // 5, 6
        i0 = nn / HWp;                                 // b
        int rem = nn % HWp;
        i1 = rem / WP;                                 // oy
        i2 = rem % WP;                                 // ox
    }
}

// representative gather base address for software prefetch
template<int STAGE>
__device__ __forceinline__ const _Float16* prefetchAddr(const GP& g,
                                                        int i0, int i1, int i2) {
    if constexpr (STAGE == 5)
        return g.Bsrc + (((size_t)i0 * 32) * HP + i1) * WP + i2;
    else if constexpr (STAGE == 6)
        return g.Bsrc + (((size_t)i0 * 64) * HP + i1) * WP + i2;
    else
        return g.Bsrc + i0;
}

// ---------------- per-k gather given decoded n ----------------
template<int STAGE>
__device__ __forceinline__ _Float16 gatherK(const GP& g, int i0, int i1, int i2, int k) {
    if constexpr (STAGE == 1) {
        if (k >= 27) return (_Float16)0.0f;
        return g.Bsrc[i0 + (k / 9) * 81 + ((k % 9) / 3) * 9 + (k % 3)];
    } else if constexpr (STAGE == 2) {
        return g.Bsrc[i0 + (k / 9) * 25 + ((k % 9) / 3) * 5 + (k % 3)];
    } else if constexpr (STAGE == 3) {
        return g.Bsrc[i0 + (k / 9) * 9 + (k % 9)];
    } else if constexpr (STAGE == 5) {                 // conv3 5x5 pad2, 32ch
        int c = k / 25, rs = k % 25, r = rs / 5, s = rs % 5;
        int iy = i1 + r - 2, ix = i2 + s - 2;
        if (iy < 0 || iy >= HP || ix < 0 || ix >= WP) return (_Float16)0.0f;
        return g.Bsrc[((i0 * 32 + c) * HP + iy) * WP + ix];
    } else {                                           // conv4 3x3 pad1, 64ch
        int c = k / 9, rs = k % 9, r = rs / 3, s = rs % 3;
        int iy = i1 + r - 1, ix = i2 + s - 1;
        if (iy < 0 || iy >= HP || ix < 0 || ix >= WP) return (_Float16)0.0f;
        return g.Bsrc[((i0 * 64 + c) * HP + iy) * WP + ix];
    }
}

// ---------------- per-stage epilogue store ----------------
template<int STAGE>
__device__ __forceinline__ void storeC(const GP& g, int m, int n, float v) {
    if constexpr (STAGE == 1) {
        v = fmaxf(v + g.bias0[m], 0.0f);
        int p = n / 25, pix = n % 25;
        g.outH[(p * 64 + m) * 25 + pix] = (_Float16)v;
    } else if constexpr (STAGE == 2) {
        v = fmaxf(v + g.bias0[m], 0.0f);
        int p = n / 9, pix = n % 9;
        g.outH[(p * 64 + m) * 9 + pix] = (_Float16)v;
    } else if constexpr (STAGE == 3) {
        v += (m < 864) ? g.bias0[m] : g.bias1[m - 864];
        g.outH[n * 896 + m] = (_Float16)v;
    } else if constexpr (STAGE == 5) {
        v = fmaxf(v + g.bias0[m], 0.0f);
        int b = n / HWp, rem = n % HWp;
        g.outH[(b * 64 + m) * HWp + rem] = (_Float16)v;
    } else {                               // STAGE 6: pixel shuffle + crop + mean
        v += g.bias0[m];
        int b = n / HWp, rem = n % HWp;
        int oy = rem / WP, ox = rem % WP;
        int co = m / 4, sy = (m % 4) / 2, sx = m % 2;
        int H2 = oy * 2 + sy, W2 = ox * 2 + sx;
        if (H2 < 720 && W2 < 1280)
            g.outF[((b * 3 + co) * 720 + H2) * 1280 + W2] = v + rgb_mean(co);
    }
}

// assemble a 16-half WMMA fragment from a 32-half LDS row:
// halves 0..7  <- k = hi*8 + 0..7      (contiguous -> ds_load_b128)
// halves 8..15 <- k = 16 + hi*8 + 0..7 (contiguous -> ds_load_b128)
__device__ __forceinline__ v16h loadFrag(const _Float16* row, int hi) {
    const v8h* p = (const v8h*)row;          // 4 chunks of 8 halves per row
    v8h lo = p[hi];
    v8h hiC = p[2 + hi];
    return __builtin_shufflevector(lo, hiC, 0, 1, 2, 3, 4, 5, 6, 7,
                                            8, 9, 10, 11, 12, 13, 14, 15);
}

// TDM: DMA a [rows x 32] f16 tile (row stride = Kp elements) global -> LDS.
// D# built per CDNA5 ISA (2D tensor, groups 2/3 zero).
__device__ __forceinline__ void tdm_load_tile(const _Float16* gsrc, void* ldsDst,
                                              int rows, int kp) {
    unsigned long long ga = (unsigned long long)(uintptr_t)gsrc;
    unsigned int ldsAddr = (unsigned int)(uintptr_t)ldsDst;   // addr[31:0] = LDS offset
    u32x4 g0;
    g0[0] = 1u;                                   // count=1, user mode, no gather
    g0[1] = ldsAddr;                              // lds_addr
    g0[2] = (unsigned int)ga;                     // global_addr[31:0]
    g0[3] = (unsigned int)((ga >> 32) & 0x1FFFFFFu) | (2u << 30);  // addr[56:32], type=2
    unsigned int td0 = (unsigned int)kp;          // tensor_dim0 (elements)
    unsigned int td1 = 1024u;                     // tensor_dim1 (>= any tile rows)
    unsigned int tile0 = 32u, tile1 = (unsigned int)rows;
    unsigned long long st0 = (unsigned long long)kp;  // tensor_dim0_stride
    i32x8 g1;
    g1[0] = 0x00010000;                           // wg_mask=0, data_size=1 (2 bytes)
    g1[1] = (int)((td0 & 0xFFFFu) << 16);         // [63:48] = tensor_dim0 lo16
    g1[2] = (int)((td0 >> 16) | ((td1 & 0xFFFFu) << 16));
    g1[3] = (int)((td1 >> 16) | (tile0 << 16));   // tile_dim0
    g1[4] = (int)tile1;                           // tile_dim1, tile_dim2=0
    g1[5] = (int)(unsigned int)st0;               // stride0[31:0]
    g1[6] = (int)(unsigned int)((st0 >> 32) & 0xFFFFu);  // stride0[47:32], stride1 lo=0
    g1[7] = 0;
    i32x4 z4 = {0, 0, 0, 0};
#if defined(__clang_major__) && __clang_major__ >= 23
    i32x8 z8 = {0, 0, 0, 0, 0, 0, 0, 0};
    __builtin_amdgcn_tensor_load_to_lds(g0, g1, z4, z4, z8, 0);
#else
    __builtin_amdgcn_tensor_load_to_lds(g0, g1, z4, z4, 0);
#endif
}

// ---------------- WMMA implicit GEMM ----------------
// 8 waves/block; wave grid WAM x WAN; each wave owns a (TM*16) x (TN*16) C tile.
template<int STAGE, int WAM, int WAN, int TM, int TN>
__global__ __launch_bounds__(256) void gemm_wmma(GP g) {
    constexpr int BM = WAM * TM * 16, BN = WAN * TN * 16;
    constexpr int KC = (32 * BN) / 256;      // k-span per thread for B fill
    __shared__ __align__(128) _Float16 lA[BM * 32];   // [m][k]
    __shared__ __align__(128) _Float16 lBt[BN * 32];  // transposed [n][k]

    const int tid  = threadIdx.x;
    const int wave = tid >> 5;
    const int lane = tid & 31;
    const int hi   = lane >> 4;
    const int lr   = lane & 15;
    const int wm   = wave % WAM;
    const int wn   = wave / WAM;
    const int mBlock = blockIdx.y * BM;
    const int nBlock = blockIdx.x * BN;

    // B-fill assignment: threads 0..BN-1 take consecutive n (coalesced gathers),
    // upper thread groups take higher k sub-ranges of the same columns.
    const int fn = tid % BN;
    const int fk = (tid / BN) * KC;
    int i0, i1, i2; bool nval;
    decodeN<STAGE>(g, nBlock + fn, i0, i1, i2, nval);
    if (nval) __builtin_prefetch(prefetchAddr<STAGE>(g, i0, i1, i2), 0, 0);

    v8f acc[TM][TN];
#pragma unroll
    for (int a = 0; a < TM; ++a)
#pragma unroll
        for (int b = 0; b < TN; ++b) acc[a][b] = (v8f)0.0f;

    const int nK = g.Kp >> 5;
    for (int kt = 0; kt < nK; ++kt) {
        const int k0 = kt << 5;
        // A tile: Tensor Data Mover DMA (wave 0 only; TDM ignores EXEC)
        if (wave == 0) {
            tdm_load_tile(g.A + (size_t)mBlock * g.Kp + k0, lA, BM, g.Kp);
            __builtin_amdgcn_s_wait_tensorcnt(0);
        }
        // B tile: im2col gather into transposed layout
#pragma unroll
        for (int e = 0; e < KC; ++e) {
            int kk = fk + e;
            _Float16 v = nval ? gatherK<STAGE>(g, i0, i1, i2, k0 + kk)
                              : (_Float16)0.0f;
            lBt[fn * 32 + kk] = v;
        }
        __syncthreads();

        v16h af[TM], bf[TN];
#pragma unroll
        for (int a = 0; a < TM; ++a)
            af[a] = loadFrag(&lA[((wm * TM + a) * 16 + lr) * 32], hi);
#pragma unroll
        for (int b = 0; b < TN; ++b)
            bf[b] = loadFrag(&lBt[((wn * TN + b) * 16 + lr) * 32], hi);
#pragma unroll
        for (int a = 0; a < TM; ++a)
#pragma unroll
            for (int b = 0; b < TN; ++b)
                acc[a][b] = __builtin_amdgcn_wmma_f32_16x16x32_f16(
                    false, af[a], false, bf[b], (short)0, acc[a][b], false, false);
        __syncthreads();
    }

    // epilogue: C layout vgpr j -> M = j + hi*8, N = lr
#pragma unroll
    for (int a = 0; a < TM; ++a) {
        const int mBase = mBlock + (wm * TM + a) * 16 + hi * 8;
#pragma unroll
        for (int b = 0; b < TN; ++b) {
            const int nIdx = nBlock + (wn * TN + b) * 16 + lr;
#pragma unroll
            for (int j = 0; j < 8; ++j) {
                int m = mBase + j;
                if (m < g.Mreal && nIdx < g.N) storeC<STAGE>(g, m, nIdx, acc[a][b][j]);
            }
        }
    }
}

// ---------------- helper kernels ----------------
__global__ void prep_weight(const float* __restrict__ src, _Float16* __restrict__ dst,
                            int M, int K, int Kp, int rowOff, int total) {
    int t = blockIdx.x * 256 + threadIdx.x;
    if (t >= total) return;
    int m = t / Kp, k = t % Kp;
    float v = (m < M && k < K) ? src[m * K + k] : 0.0f;
    dst[(rowOff + m) * Kp + k] = (_Float16)v;
}

// build mean-subtracted, zero-bordered 9x9 patches (Bp,3,9,9) f16
__global__ void build_patches(const float* __restrict__ x, _Float16* __restrict__ patches9) {
    int t = blockIdx.x * 256 + threadIdx.x;
    if (t >= BP * 243) return;
    int idx = t % 81, c = (t / 81) % 3, p = t / 243;
    int i = idx / 9, j = idx % 9;
    _Float16 v = (_Float16)0.0f;
    if (i >= 1 && i <= 7 && j >= 1 && j <= 7) {
        int b = p / (GH * GW), pg = p % (GH * GW);
        int gy = pg / GW, gx = pg % GW;
        int Y = gy * 7 + i - 1, X = gx * 7 + j - 1;
        if (Y < H0 && X < W0)
            v = (_Float16)(x[((b * 3 + c) * H0 + Y) * W0 + X] - rgb_mean(c));
    }
    patches9[t] = v;
}

// per-patch predicted-kernel application; writes reassembled (NB,32,HP,WP) layout
__global__ void einsum_apply(const _Float16* __restrict__ kernbuf,
                             const _Float16* __restrict__ patches9,
                             _Float16* __restrict__ ybuf) {
    int t = blockIdx.x * 256 + threadIdx.x;
    if (t >= BP * 32 * 49) return;
    int pix = t % 49, f = (t / 49) % 32, p = t / (49 * 32);
    int h = pix / 7, w = pix % 7;
    float acc = (float)kernbuf[p * 896 + 864 + f];   // bias term (bb already added)
    const _Float16* kr = kernbuf + p * 896 + f * 27;
    const _Float16* pp = patches9 + p * 243;
#pragma unroll
    for (int c = 0; c < 3; ++c)
#pragma unroll
        for (int r = 0; r < 3; ++r)
#pragma unroll
            for (int s = 0; s < 3; ++s)
                acc += (float)kr[c * 9 + r * 3 + s] *
                       (float)pp[c * 81 + (h + r) * 9 + (w + s)];
    int b = p / (GH * GW), pg = p % (GH * GW);
    int gy = pg / GW, gx = pg % GW;
    ybuf[((b * 32 + f) * HP + gy * 7 + h) * WP + gx * 7 + w] = (_Float16)acc;
}

// ---------------- host side ----------------
static inline size_t cdiv(size_t a, size_t b) { return (a + b - 1) / b; }

extern "C" void kernel_launch(void* const* d_in, const int* in_sizes, int n_in,
                              void* d_out, int out_size, void* d_ws, size_t ws_size,
                              hipStream_t stream) {
    (void)in_sizes; (void)n_in; (void)out_size; (void)ws_size;
    const float* x  = (const float*)d_in[0];
    const float* w1 = (const float*)d_in[1];  const float* b1 = (const float*)d_in[2];
    const float* w2 = (const float*)d_in[3];  const float* b2 = (const float*)d_in[4];
    const float* wk = (const float*)d_in[5];  const float* bk = (const float*)d_in[6];
    const float* wb = (const float*)d_in[7];  const float* bb = (const float*)d_in[8];
    const float* w3 = (const float*)d_in[9];  const float* b3 = (const float*)d_in[10];
    const float* w4 = (const float*)d_in[11]; const float* b4 = (const float*)d_in[12];
    float* out = (float*)d_out;

    // workspace carve-out (all f16), 256B aligned
    char* base = (char*)d_ws;
    size_t off = 0;
    auto carve = [&](size_t halves) {
        char* p = base + off;
        off = (off + halves * sizeof(_Float16) + 255) & ~(size_t)255;
        return (_Float16*)p;
    };
    _Float16* A1       = carve((size_t)64 * 32);
    _Float16* A2       = carve((size_t)64 * 576);
    _Float16* A3       = carve((size_t)896 * 576);
    _Float16* A5       = carve((size_t)64 * 800);
    _Float16* A6       = carve((size_t)16 * 576);
    _Float16* patches9 = carve((size_t)BP * 243);
    _Float16* feat1    = carve((size_t)BP * 64 * 25);
    _Float16* feat2    = carve((size_t)BP * 64 * 9);
    _Float16* kernbuf  = carve((size_t)BP * 896);
    _Float16* ybuf     = carve((size_t)NB * 32 * HWp);
    _Float16* c3out    = carve((size_t)NB * 64 * HWp);

    dim3 blk(256);
    // ---- weight prep (f32 -> padded f16 GEMM-A matrices) ----
    prep_weight<<<cdiv(64 * 32, 256),  blk, 0, stream>>>(w1, A1, 64, 27, 32, 0, 64 * 32);
    prep_weight<<<cdiv(64 * 576, 256), blk, 0, stream>>>(w2, A2, 64, 576, 576, 0, 64 * 576);
    prep_weight<<<cdiv(864 * 576, 256), blk, 0, stream>>>(wk, A3, 864, 576, 576, 0, 864 * 576);
    prep_weight<<<cdiv(32 * 576, 256), blk, 0, stream>>>(wb, A3, 32, 576, 576, 864, 32 * 576);
    prep_weight<<<cdiv(64 * 800, 256), blk, 0, stream>>>(w3, A5, 64, 800, 800, 0, 64 * 800);
    prep_weight<<<cdiv(16 * 576, 256), blk, 0, stream>>>(w4, A6, 12, 576, 576, 0, 16 * 576);

    // ---- patch extraction ----
    build_patches<<<cdiv((size_t)BP * 243, 256), blk, 0, stream>>>(x, patches9);

    // ---- stage 1: conv1 (M=64,K=32,N=Bp*25), block 64x128, wave 32x32 ----
    {
        GP g{A1, patches9, feat1, nullptr, b1, b1, 64, 64, 32, N1};
        gemm_wmma<1, 2, 4, 2, 2><<<dim3((unsigned)cdiv(N1, 128), 1), blk, 0, stream>>>(g);
    }
    // ---- stage 2: conv2 (M=64,K=576,N=Bp*9) ----
    {
        GP g{A2, feat1, feat2, nullptr, b2, b2, 64, 64, 576, N2};
        gemm_wmma<2, 2, 4, 2, 2><<<dim3((unsigned)cdiv(N2, 128), 1), blk, 0, stream>>>(g);
    }
    // ---- stage 3: fused kern+bias GEMM (M=896,K=576,N=Bp) ----
    {
        GP g{A3, feat2, kernbuf, nullptr, bk, bb, 896, 896, 576, N3};
        gemm_wmma<3, 2, 4, 2, 2><<<dim3((unsigned)cdiv(N3, 128), 896 / 64), blk, 0, stream>>>(g);
    }
    // ---- stage 4: per-patch kernel application + reassembly ----
    einsum_apply<<<cdiv((size_t)BP * 32 * 49, 256), blk, 0, stream>>>(kernbuf, patches9, ybuf);

    // ---- stage 5: conv3 5x5 pad2 (M=64,K=800,N=937664) ----
    {
        GP g{A5, ybuf, c3out, nullptr, b3, b3, 64, 64, 800, N56};
        gemm_wmma<5, 2, 4, 2, 2><<<dim3((unsigned)cdiv(N56, 128), 1), blk, 0, stream>>>(g);
    }
    // ---- stage 6: conv4 3x3 pad1 + pixel shuffle (M=16,K=576), block 16x256 ----
    {
        GP g{A6, c3out, nullptr, out, b4, b4, 12, 16, 576, N56};
        gemm_wmma<6, 1, 8, 1, 2><<<dim3((unsigned)cdiv(N56, 256), 1), blk, 0, stream>>>(g);
    }
}
